// Hecto_16595753631800
// MI455X (gfx1250) — matrix operand order repaired and checked by reading, not compiled, not run on hardware
//
#include <hip/hip_runtime.h>
#include <hip/hip_bf16.h>
#include <cstddef>

typedef __attribute__((ext_vector_type(16))) __bf16 v16bf;
typedef __attribute__((ext_vector_type(8)))  float  v8f;
typedef __attribute__((ext_vector_type(4)))  float  f4;

static __device__ __forceinline__ v8f wmma_bf16(v16bf a, v16bf b, v8f c) {
    // D = A(16x32 bf16) * B(32x16 bf16) + C(16x16 f32)
    return __builtin_amdgcn_wmma_f32_16x16x32_bf16(
        /*neg_a=*/false, a, /*neg_b=*/false, b,
        /*c_mod=*/(short)0, c, /*reuse_a=*/false, /*reuse_b=*/false);
}

// ---------------------------------------------------------------------------
// Kernel 1: global average pool of one feature scale.
// feat: [64,256,s,s] row-major -> pooled[b][scale][c], pooled is [64][4][256].
// One 256-thread block per (b,c) row; streaming non-temporal b128 loads.
// ---------------------------------------------------------------------------
__global__ void feat_pool_kernel(const float* __restrict__ feat,
                                 float* __restrict__ pooled,
                                 int spatial, int scale) {
    int bc  = blockIdx.x;            // b*256 + c
    int tid = threadIdx.x;
    const f4* p4 = (const f4*)(feat + (size_t)bc * spatial);
    int n4 = spatial >> 2;
    float s = 0.f;
    for (int i = tid; i < n4; i += 256) {
        f4 v = __builtin_nontemporal_load(p4 + i);
        s += v.x + v.y + v.z + v.w;
    }
    __shared__ float red[256];
    red[tid] = s;
    __syncthreads();
    for (int off = 128; off > 0; off >>= 1) {
        if (tid < off) red[tid] += red[tid + off];
        __syncthreads();
    }
    if (tid == 0) {
        int b = bc >> 8, c = bc & 255;
        pooled[((size_t)b * 4 + scale) * 256 + c] = red[0] / (float)spatial;
    }
}

// ---------------------------------------------------------------------------
// Kernel 2: per-class masked mean pooling over queries.
// pred_logits [64,300,6], hs [64,300,256] -> tokens[b][0..5][256], validf[b][0..9]
// sigmoid(x) > 0.3  <=>  x > ln(0.3/0.7)
// ---------------------------------------------------------------------------
__global__ void class_pool_kernel(const float* __restrict__ pred_logits,
                                  const float* __restrict__ hs,
                                  float* __restrict__ tokens,   // [64,10,256]
                                  float* __restrict__ validf) { // [64,10]
    __shared__ unsigned int maskbits[300];
    __shared__ float counts[6];
    __shared__ int s_any;
    const float LT = -0.84729786038720367f;   // ln(3/7)
    int b = blockIdx.x, tid = threadIdx.x;

    for (int q = tid; q < 300; q += 256) {
        const float* pl = pred_logits + ((size_t)b * 300 + q) * 6;
        unsigned int mb = 0;
        #pragma unroll
        for (int p = 0; p < 6; ++p) if (pl[p] > LT) mb |= (1u << p);
        maskbits[q] = mb;
    }
    __syncthreads();
    if (tid < 6) {
        float cnt = 0.f;
        for (int q = 0; q < 300; ++q) cnt += (float)((maskbits[q] >> tid) & 1u);
        counts[tid] = cnt;
    }
    __syncthreads();
    if (tid == 0) {
        int any = 0;
        for (int p = 0; p < 6; ++p) if (counts[p] > 0.f) any = 1;
        s_any = any;
    }
    __syncthreads();

    const float* hsb = hs + (size_t)b * 300 * 256;
    float acc[6] = {0.f, 0.f, 0.f, 0.f, 0.f, 0.f};
    for (int q = 0; q < 300; ++q) {
        float v = hsb[(size_t)q * 256 + tid];
        unsigned int mb = maskbits[q];
        #pragma unroll
        for (int p = 0; p < 6; ++p) if (mb & (1u << p)) acc[p] += v;
    }
    bool anyv = (s_any != 0);
    #pragma unroll
    for (int p = 0; p < 6; ++p) {
        float mean = acc[p] / fmaxf(counts[p], 1.0f);
        if (p == 0 && !anyv) mean = hsb[tid];  // fallback: hs[b,0,:]
        tokens[((size_t)b * 10 + p) * 256 + tid] = mean;
    }
    if (tid < 10) {
        float v;
        if (tid < 6) {
            v = (counts[tid] > 0.f) ? 1.f : 0.f;
            if (tid == 0 && !anyv) v = 1.f;
        } else v = 1.f;  // image tokens always valid
        validf[(size_t)b * 10 + tid] = v;
    }
}

// ---------------------------------------------------------------------------
// Generic WMMA GEMM: out[M,N] = (relu?)(A[M,K](lda) @ W[N,K]^T + bias)
// One wave per 16x16 output tile; A loaded straight from global.
// Fragment layouts per CDNA5 ISA 7.12.2 (16-bit A 16x32, 16-bit B 32x16).
// ---------------------------------------------------------------------------
__global__ void gemm_wmma_kernel(const float* __restrict__ A, int lda,
                                 const float* __restrict__ W,  // [N,K]
                                 const float* __restrict__ bias,
                                 float* __restrict__ out, int ldo,
                                 int M, int N, int K, int relu) {
    int lane = threadIdx.x & 31;
    int wave = threadIdx.x >> 5;
    int tilesN = (N + 15) >> 4;
    int tilesM = M >> 4;                       // M multiple of 16 by construction
    int gid = blockIdx.x * (blockDim.x >> 5) + wave;
    if (gid >= tilesM * tilesN) return;
    int tm = gid / tilesN, tn = gid % tilesN;
    int l15 = lane & 15, kh = lane >> 4;

    const float* Ap = A + (size_t)(tm * 16 + l15) * lda;
    int wrow = tn * 16 + l15; if (wrow >= N) wrow = N - 1;   // clamp (N=396 case)
    const float* Wp = W + (size_t)wrow * K;

    v8f c = {};
    for (int kc = 0; kc < K; kc += 32) {
        v16bf a, bb;
        #pragma unroll
        for (int j = 0; j < 8; ++j) {
            int ka = kc + ((j & 3) << 1) + ((j >> 2) << 4) + (kh << 3);
            a[2 * j]     = (__bf16)Ap[ka];
            a[2 * j + 1] = (__bf16)Ap[ka + 1];
            int kb = kc + ((j & 3) << 1) + ((j >> 2) << 4) + (kh << 3);
            bb[2 * j]     = (__bf16)Wp[kb];
            bb[2 * j + 1] = (__bf16)Wp[kb + 1];
        }
        c = wmma_bf16(a, bb, c);
    }
    int ncol = tn * 16 + l15;
    if (ncol < N) {
        float bv = bias ? bias[ncol] : 0.f;
        #pragma unroll
        for (int r = 0; r < 8; ++r) {
            int m = tm * 16 + r + kh * 8;      // C/D layout: upper lanes M+=8
            float v = c[r] + bv;
            if (relu) v = fmaxf(v, 0.f);
            out[(size_t)m * ldo + ncol] = v;
        }
    }
}

// ---------------------------------------------------------------------------
// WMMA GEMM with A in LDS (M=16 fixed): D[16,N](ldd, LDS) = A[16,K](lda, LDS)
// @ W[N,K]^T + bias.  Each of the 8 waves takes N-tiles round-robin.
// ---------------------------------------------------------------------------
static __device__ __forceinline__ void gemm_lds(
        const float* __restrict__ Alds, int lda,
        const float* __restrict__ W, const float* __restrict__ bias,
        float* __restrict__ Dlds, int ldd,
        int Ntiles, int K, int relu, int lane, int wave) {
    int l15 = lane & 15, kh = lane >> 4;
    const float* Ap = Alds + (size_t)l15 * lda;
    for (int tn = wave; tn < Ntiles; tn += 8) {
        const float* Wp = W + (size_t)(tn * 16 + l15) * K;
        v8f c = {};
        for (int kc = 0; kc < K; kc += 32) {
            v16bf a, bb;
            #pragma unroll
            for (int j = 0; j < 8; ++j) {
                int ka = kc + ((j & 3) << 1) + ((j >> 2) << 4) + (kh << 3);
                a[2 * j]     = (__bf16)Ap[ka];
                a[2 * j + 1] = (__bf16)Ap[ka + 1];
                int kb = kc + ((j & 3) << 1) + ((j >> 2) << 4) + (kh << 3);
                bb[2 * j]     = (__bf16)Wp[kb];
                bb[2 * j + 1] = (__bf16)Wp[kb + 1];
            }
            c = wmma_bf16(a, bb, c);
        }
        int ncol = tn * 16 + l15;
        float bv = bias[ncol];
        #pragma unroll
        for (int r = 0; r < 8; ++r) {
            float v = c[r] + bv;
            if (relu) v = fmaxf(v, 0.f);
            Dlds[(size_t)(r + kh * 8) * ldd + ncol] = v;
        }
    }
}

// x[0..9] := LayerNorm(x + res) * w + b   (rows 10..15 of x stay zero)
static __device__ __forceinline__ void add_ln(float* x, const float* res,
                                              const float* w, const float* bb,
                                              float* st, int tid) {
    for (int i = tid; i < 2560; i += 256) x[i] += res[i];
    __syncthreads();
    if (tid < 10) {
        const float* row = x + tid * 256;
        float s = 0.f, s2 = 0.f;
        for (int d = 0; d < 256; ++d) { float v = row[d]; s += v; s2 += v * v; }
        float m = s * (1.f / 256.f);
        st[tid]      = m;
        st[16 + tid] = s2 * (1.f / 256.f) - m * m;
    }
    __syncthreads();
    for (int i = tid; i < 2560; i += 256) {
        int r = i >> 8, d = i & 255;
        float inv = rsqrtf(st[16 + r] + 1e-5f);
        x[i] = (x[i] - st[r]) * inv * w[d] + bb[d];
    }
    __syncthreads();
}

// ---------------------------------------------------------------------------
// Kernel 3: 2-layer post-norm transformer encoder over 10 tokens per batch.
// One workgroup (8 wave32) per batch element; all projections via WMMA.
// ---------------------------------------------------------------------------
#define TR_SMEM_FLOATS (4096 + 12288 + 4096 + 4096 + 800 + 48)
__global__ void transformer_kernel(
        const float* __restrict__ tokens, const float* __restrict__ validf,
        const float* __restrict__ in_w,  const float* __restrict__ in_b,
        const float* __restrict__ ow,    const float* __restrict__ ob,
        const float* __restrict__ l1w,   const float* __restrict__ l1b,
        const float* __restrict__ l2w,   const float* __restrict__ l2b,
        const float* __restrict__ ln1w,  const float* __restrict__ ln1b,
        const float* __restrict__ ln2w,  const float* __restrict__ ln2b,
        float* __restrict__ cls_tok) {
    extern __shared__ float sm[];
    float* x    = sm;                 // [16][256] tokens (rows 10..15 = 0)
    float* bufA = x    + 4096;        // [16][768] qkv  /  [16][512] ff1
    float* bufB = bufA + 12288;       // [16][256] attention output
    float* bufC = bufB + 4096;        // [16][256] projection result
    float* sc   = bufC + 4096;        // [8][10][10] attention scores
    float* st   = sc   + 800;         // mean[16], var[16], valid[10]

    int b = blockIdx.x, tid = threadIdx.x;
    int lane = tid & 31, wave = tid >> 5;

    for (int i = tid; i < 16 * 256; i += 256) {
        int m = i >> 8;
        x[i] = (m < 10) ? tokens[((size_t)b * 10 + m) * 256 + (i & 255)] : 0.f;
    }
    if (tid < 10) st[32 + tid] = validf[(size_t)b * 10 + tid];
    __syncthreads();

    const float scl = 0.17677669529663689f;  // 1/sqrt(32)
    for (int l = 0; l < 2; ++l) {
        // qkv = x @ in_w[l].T + in_b[l]   -> bufA [16][768]
        gemm_lds(x, 256, in_w + (size_t)l * 768 * 256, in_b + l * 768,
                 bufA, 768, 48, 256, 0, lane, wave);
        __syncthreads();
        // scores[h][i][j] (+ key-padding bias)
        for (int idx = tid; idx < 800; idx += 256) {
            int h = idx / 100, rem = idx % 100, i = rem / 10, j = rem % 10;
            const float* qi = bufA + (size_t)i * 768 + h * 32;
            const float* kj = bufA + (size_t)j * 768 + 256 + h * 32;
            float s = 0.f;
            #pragma unroll
            for (int d = 0; d < 32; ++d) s += qi[d] * kj[d];
            float kb = (st[32 + j] > 0.5f) ? 0.f : -1e30f;
            sc[idx] = s * scl + kb;
        }
        __syncthreads();
        // softmax over j for each of the 80 (h,i) rows
        if (tid < 80) {
            float* row = sc + tid * 10;
            float mx = row[0];
            for (int j = 1; j < 10; ++j) mx = fmaxf(mx, row[j]);
            float sum = 0.f;
            for (int j = 0; j < 10; ++j) { float e = __expf(row[j] - mx); row[j] = e; sum += e; }
            float inv = 1.f / sum;
            for (int j = 0; j < 10; ++j) row[j] *= inv;
        }
        __syncthreads();
        // ao = attn @ v   -> bufB [16][256] (pad rows zeroed)
        for (int idx = tid; idx < 16 * 256; idx += 256) {
            int i = idx >> 8, dd = idx & 255;
            float v = 0.f;
            if (i < 10) {
                int h = dd >> 5;
                const float* arow = sc + (size_t)(h * 10 + i) * 10;
                const float* vcol = bufA + 512 + dd;
                #pragma unroll
                for (int j = 0; j < 10; ++j) v += arow[j] * vcol[(size_t)j * 768];
            }
            bufB[idx] = v;
        }
        __syncthreads();
        // out-proj -> bufC; x = LN1(x + bufC)
        gemm_lds(bufB, 256, ow + (size_t)l * 256 * 256, ob + l * 256,
                 bufC, 256, 16, 256, 0, lane, wave);
        __syncthreads();
        add_ln(x, bufC, ln1w + l * 256, ln1b + l * 256, st, tid);
        // ff1 = relu(x @ l1w.T + b) -> bufA [16][512]
        gemm_lds(x, 256, l1w + (size_t)l * 512 * 256, l1b + l * 512,
                 bufA, 512, 32, 256, 1, lane, wave);
        __syncthreads();
        // ff2 = ff1 @ l2w.T + b -> bufC; x = LN2(x + bufC)
        gemm_lds(bufA, 512, l2w + (size_t)l * 256 * 512, l2b + l * 256,
                 bufC, 256, 16, 512, 0, lane, wave);
        __syncthreads();
        add_ln(x, bufC, ln2w + l * 256, ln2b + l * 256, st, tid);
    }
    // masked mean pool over valid tokens -> cls_tok[b][256]
    float len = 0.f;
    for (int m = 0; m < 10; ++m) len += st[32 + m];
    len = fmaxf(len, 1.0f);
    float s = 0.f;
    for (int m = 0; m < 10; ++m) s += x[m * 256 + tid] * st[32 + m];
    cls_tok[(size_t)b * 256 + tid] = s / len;
}

// ---------------------------------------------------------------------------
extern "C" void kernel_launch(void* const* d_in, const int* in_sizes, int n_in,
                              void* d_out, int out_size, void* d_ws, size_t ws_size,
                              hipStream_t stream) {
    const float* pred   = (const float*)d_in[0];
    const float* hs     = (const float*)d_in[1];
    const float* feats[4] = {(const float*)d_in[2], (const float*)d_in[3],
                             (const float*)d_in[4], (const float*)d_in[5]};
    const float* img_w  = (const float*)d_in[6];
    const float* img_b  = (const float*)d_in[7];
    const float* in_w   = (const float*)d_in[8];
    const float* in_b   = (const float*)d_in[9];
    const float* ow     = (const float*)d_in[10];
    const float* ob     = (const float*)d_in[11];
    const float* l1w    = (const float*)d_in[12];
    const float* l1b    = (const float*)d_in[13];
    const float* l2w    = (const float*)d_in[14];
    const float* l2b    = (const float*)d_in[15];
    const float* ln1w   = (const float*)d_in[16];
    const float* ln1b   = (const float*)d_in[17];
    const float* ln2w   = (const float*)d_in[18];
    const float* ln2b   = (const float*)d_in[19];
    const float* cls1_w = (const float*)d_in[20];
    const float* cls1_b = (const float*)d_in[21];
    const float* cls2_w = (const float*)d_in[22];
    const float* cls2_b = (const float*)d_in[23];
    float* out = (float*)d_out;

    float* ws      = (float*)d_ws;
    float* pooled  = ws;                       // [64][4][256]   = 65536
    float* tokens  = pooled  + 64 * 4 * 256;   // [64][10][256]  = 163840
    float* validf  = tokens  + 64 * 10 * 256;  // [64][10]       = 640
    float* cls_tok = validf  + 640;            // [64][256]      = 16384
    float* hbuf    = cls_tok + 64 * 256;       // [64][512]      = 32768

    // 1) streaming global-average pool of the 4 feature scales (HBM-bound)
    const int spat[4] = {96 * 96, 48 * 48, 24 * 24, 12 * 12};
    for (int i = 0; i < 4; ++i)
        feat_pool_kernel<<<64 * 256, 256, 0, stream>>>(feats[i], pooled, spat[i], i);

    // 2) per-class masked mean pooling of hs
    class_pool_kernel<<<64, 256, 0, stream>>>(pred, hs, tokens, validf);

    // 3) per-scale image token projection (WMMA): tokens[:,6+i,:] = pooled[:,i,:] @ W_i^T + b_i
    for (int i = 0; i < 4; ++i)
        gemm_wmma_kernel<<<8, 256, 0, stream>>>(pooled + i * 256, 1024,
                                                img_w + (size_t)i * 256 * 256,
                                                img_b + i * 256,
                                                tokens + (6 + i) * 256, 2560,
                                                64, 256, 256, 0);

    // 4) transformer encoder (WMMA projections) + masked mean pool
    transformer_kernel<<<64, 256, TR_SMEM_FLOATS * sizeof(float), stream>>>(
        tokens, validf, in_w, in_b, ow, ob, l1w, l1b, l2w, l2b,
        ln1w, ln1b, ln2w, ln2b, cls_tok);

    // 5) classifier: h = relu(cls_tok @ cls1^T + b1); logits = h @ cls2^T + b2
    gemm_wmma_kernel<<<16, 256, 0, stream>>>(cls_tok, 256, cls1_w, cls1_b,
                                             hbuf, 512, 64, 512, 256, 1);
    // tiles = 4 * ceil(396/16)=25 -> 100 waves -> 13 blocks
    gemm_wmma_kernel<<<13, 256, 0, stream>>>(hbuf, 512, cls2_w, cls2_b,
                                             out, 396, 64, 396, 512, 0);
}
